// LocalAttention_85564338471273
// MI455X (gfx1250) — compile-verified, matrix operand order checked by reading, not run on hardware
//
#include <hip/hip_runtime.h>

typedef __attribute__((ext_vector_type(16))) __bf16       v16bf;
typedef __attribute__((ext_vector_type(8)))  float        v8f;
typedef __attribute__((ext_vector_type(8)))  unsigned int v8u;

#define TSEQ 8192
#define DH   64
#define WSZ  128
#define NWIN 64   // TSEQ / WSZ

// ---- helpers -------------------------------------------------------------
static __device__ __forceinline__ unsigned int pk2(float a, float b) {
    unsigned short ua = __builtin_bit_cast(unsigned short, (__bf16)a);
    unsigned short ub = __builtin_bit_cast(unsigned short, (__bf16)b);
    return (unsigned int)ua | ((unsigned int)ub << 16);
}
static __device__ __forceinline__ float tobf(float a) { return (float)(__bf16)a; }

static __device__ __forceinline__ v8f wmma_bf16(v16bf a, v16bf b, v8f c) {
    // D = A(16x32 bf16) * B(32x16 bf16) + C(16x16 f32)
    return __builtin_amdgcn_wmma_f32_16x16x32_bf16(false, a, false, b, (short)0, c,
                                                   false, false);
}

// LDS layout (uint units):
//  Kh / Kl : [256 keys][33]  (32 data u32 = 64 bf16 dims, +1 pad)   8448 each
//  VTh/VTl : [64 dims][129]  (128 data u32 = 256 bf16 keys, +1 pad) 8256 each
//  P       : 8 waves * 16 rows * 17 u32 (32 bf16 + 2 pad)           2176
#define KH_OFF  0
#define KL_OFF  8448
#define VTH_OFF (2 * 8448)
#define VTL_OFF (2 * 8448 + 8256)
#define P_OFF   (2 * 8448 + 2 * 8256)
#define SMEM_U32 (2 * 8448 + 2 * 8256 + 8 * 272)

__global__ void __launch_bounds__(256)
local_attn_wmma_kernel(const float* __restrict__ qin,
                       const float* __restrict__ kin,
                       const float* __restrict__ vin,
                       float* __restrict__ outp) {
    __shared__ unsigned int smem[SMEM_U32];
    unsigned int* Kh  = smem + KH_OFF;
    unsigned int* Kl  = smem + KL_OFF;
    unsigned int* VTh = smem + VTH_OFF;
    unsigned int* VTl = smem + VTL_OFF;

    const int tid  = threadIdx.x;
    const int bh   = blockIdx.x >> 6;    // 0..31
    const int w    = blockIdx.x & 63;    // 0..63 window index

    // ---- Global -> LDS: K (row major, split) and V (transposed, split) ----
    {
        const float* kbase = kin + ((size_t)bh * TSEQ + ((size_t)w * WSZ - WSZ)) * DH;
        const float* vbase = vin + ((size_t)bh * TSEQ + ((size_t)w * WSZ - WSZ)) * DH;
        __bf16* VThb = (__bf16*)VTh;   // stride 258 bf16 per dim row
        __bf16* VTlb = (__bf16*)VTl;
        for (int e = tid; e < 256 * 32; e += 256) {   // float2 granules
            const int key = e >> 5;
            const int dp  = e & 31;                    // dim pair index
            const bool valid = (w > 0) || (key >= 128);
            float2 xk = {0.f, 0.f}, xv = {0.f, 0.f};
            if (valid) {
                xk = *(const float2*)(kbase + (size_t)key * DH + dp * 2);
                xv = *(const float2*)(vbase + (size_t)key * DH + dp * 2);
            }
            const int ki = key * 33 + dp;
            Kh[ki] = pk2(xk.x, xk.y);
            Kl[ki] = pk2(xk.x - tobf(xk.x), xk.y - tobf(xk.y));
            const int d0 = dp * 2;
            VThb[(d0 + 0) * 258 + key] = (__bf16)xv.x;
            VThb[(d0 + 1) * 258 + key] = (__bf16)xv.y;
            VTlb[(d0 + 0) * 258 + key] = (__bf16)(xv.x - tobf(xv.x));
            VTlb[(d0 + 1) * 258 + key] = (__bf16)(xv.y - tobf(xv.y));
        }
    }
    __syncthreads();

    // ---- per-wave setup ---------------------------------------------------
    const int lane = tid & 31;
    const int wv   = tid >> 5;        // wave 0..7
    const int m0   = wv * 16;         // this wave's 16 query rows
    const int n    = lane & 15;
    const int half = lane >> 4;

    // Q strip in registers, A-matrix layout, scale folded with log2(e)
    const float SC = 0.125f * 1.4426950408889634f; // D^-0.5 * log2(e)
    const float* qg = qin + ((size_t)bh * TSEQ + (size_t)w * WSZ + m0 + n) * DH;
    v16bf qh[2], ql[2];
    #pragma unroll
    for (int ks = 0; ks < 2; ++ks) {
        v8u uh, ul;
        #pragma unroll
        for (int j = 0; j < 8; ++j) {
            const int kd = ks * 32 + (j >> 2) * 16 + half * 8 + (j & 3) * 2;
            float2 x = *(const float2*)(qg + kd);
            const float a = x.x * SC, b = x.y * SC;
            uh[j] = pk2(a, b);
            ul[j] = pk2(a - tobf(a), b - tobf(b));
        }
        qh[ks] = __builtin_bit_cast(v16bf, uh);
        ql[ks] = __builtin_bit_cast(v16bf, ul);
    }

    unsigned int* Pw  = smem + P_OFF + wv * 272;  // wave-private 16x32 bf16 (+pad)
    __bf16*       Pwb = (__bf16*)Pw;              // stride 34 bf16

    float mst[8], lst[8];
    v8f   Ot[4] = {};
    #pragma unroll
    for (int v = 0; v < 8; ++v) { mst[v] = -3.0e38f; lst[v] = 0.f; }

    // ---- flash loop over 8 key-blocks of 32 -------------------------------
    for (int kb = 0; kb < 8; ++kb) {
        if (w == 0 && kb < 4) continue;          // fully padded
        if (kb * 32 > 143 + m0) break;           // fully causal-masked

        // S = (Q * scale) @ K^T  (3-term bf16 split)
        v8f S[2] = {};
        #pragma unroll
        for (int ks = 0; ks < 2; ++ks) {
            #pragma unroll
            for (int nt = 0; nt < 2; ++nt) {
                v8u uh, ul;
                const int keyb = kb * 32 + nt * 16 + n;
                #pragma unroll
                for (int j = 0; j < 8; ++j) {
                    const int kd  = ks * 32 + (j >> 2) * 16 + half * 8 + (j & 3) * 2;
                    const int idx = keyb * 33 + (kd >> 1);
                    uh[j] = Kh[idx];
                    ul[j] = Kl[idx];
                }
                v16bf bhv = __builtin_bit_cast(v16bf, uh);
                v16bf blv = __builtin_bit_cast(v16bf, ul);
                S[nt] = wmma_bf16(qh[ks], bhv, S[nt]);
                S[nt] = wmma_bf16(qh[ks], blv, S[nt]);
                S[nt] = wmma_bf16(ql[ks], bhv, S[nt]);
            }
        }

        // causal + pad mask (C layout: row m = v + 8*half, col = nt*16 + n)
        #pragma unroll
        for (int nt = 0; nt < 2; ++nt) {
            const int kidx = kb * 32 + nt * 16 + n;
            const bool pad = (w == 0) && (kidx < 128);
            #pragma unroll
            for (int v = 0; v < 8; ++v) {
                const int mq = m0 + v + 8 * half;
                if (pad || (kidx > 128 + mq)) S[nt][v] = -3.0e38f;
            }
        }

        // online softmax (base-2 domain)
        float rmax[8];
        #pragma unroll
        for (int v = 0; v < 8; ++v) rmax[v] = fmaxf(S[0][v], S[1][v]);
        #pragma unroll
        for (int sh = 1; sh < 16; sh <<= 1)
            #pragma unroll
            for (int v = 0; v < 8; ++v)
                rmax[v] = fmaxf(rmax[v], __shfl_xor(rmax[v], sh));

        float corr[8];
        #pragma unroll
        for (int v = 0; v < 8; ++v) {
            const float mn = fmaxf(mst[v], rmax[v]);
            corr[v] = exp2f(mst[v] - mn);
            mst[v]  = mn;
        }
        #pragma unroll
        for (int nt = 0; nt < 2; ++nt)
            #pragma unroll
            for (int v = 0; v < 8; ++v)
                S[nt][v] = exp2f(S[nt][v] - mst[v]);

        float rsum[8];
        #pragma unroll
        for (int v = 0; v < 8; ++v) rsum[v] = S[0][v] + S[1][v];
        #pragma unroll
        for (int sh = 1; sh < 16; sh <<= 1)
            #pragma unroll
            for (int v = 0; v < 8; ++v)
                rsum[v] += __shfl_xor(rsum[v], sh);
        #pragma unroll
        for (int v = 0; v < 8; ++v) lst[v] = lst[v] * corr[v] + rsum[v];
        #pragma unroll
        for (int dt = 0; dt < 4; ++dt)
            #pragma unroll
            for (int v = 0; v < 8; ++v) Ot[dt][v] *= corr[v];

        // P: C layout -> wave-private LDS -> A layout
        #pragma unroll
        for (int nt = 0; nt < 2; ++nt)
            #pragma unroll
            for (int v = 0; v < 8; ++v)
                Pwb[(v + 8 * half) * 34 + nt * 16 + n] = (__bf16)S[nt][v];
        asm volatile("s_wait_dscnt 0" ::: "memory");
        v8u pu;
        #pragma unroll
        for (int j = 0; j < 8; ++j) {
            const int kk = (j >> 2) * 16 + half * 8 + (j & 3) * 2;
            pu[j] = Pw[n * 17 + (kk >> 1)];
        }
        v16bf pA = __builtin_bit_cast(v16bf, pu);

        // O += P @ V  (2-term split on V)
        #pragma unroll
        for (int dt = 0; dt < 4; ++dt) {
            v8u uh, ul;
            const int dim = dt * 16 + n;
            #pragma unroll
            for (int j = 0; j < 8; ++j) {
                const int kv  = (j >> 2) * 16 + half * 8 + (j & 3) * 2;
                const int idx = dim * 129 + kb * 16 + (kv >> 1);
                uh[j] = VTh[idx];
                ul[j] = VTl[idx];
            }
            Ot[dt] = wmma_bf16(pA, __builtin_bit_cast(v16bf, uh), Ot[dt]);
            Ot[dt] = wmma_bf16(pA, __builtin_bit_cast(v16bf, ul), Ot[dt]);
        }
    }

    // ---- normalize and store ----------------------------------------------
    float* og = outp + ((size_t)bh * TSEQ + (size_t)w * WSZ + m0) * DH;
    float inv[8];
    #pragma unroll
    for (int v = 0; v < 8; ++v) inv[v] = 1.0f / lst[v];
    #pragma unroll
    for (int dt = 0; dt < 4; ++dt)
        #pragma unroll
        for (int v = 0; v < 8; ++v)
            og[(size_t)(v + 8 * half) * DH + dt * 16 + n] = Ot[dt][v] * inv[v];
}

extern "C" void kernel_launch(void* const* d_in, const int* in_sizes, int n_in,
                              void* d_out, int out_size, void* d_ws, size_t ws_size,
                              hipStream_t stream) {
    const float* q = (const float*)d_in[0];
    const float* k = (const float*)d_in[1];
    const float* v = (const float*)d_in[2];
    float* out = (float*)d_out;
    (void)in_sizes; (void)n_in; (void)out_size; (void)d_ws; (void)ws_size;
    // 4*8 batch-heads * 64 windows = 2048 workgroups, 256 threads (8 wave32)
    dim3 grid(32 * NWIN), block(256);
    local_attn_wmma_kernel<<<grid, block, 0, stream>>>(q, k, v, out);
}
